// LSTMCell_59966333387333
// MI455X (gfx1250) — matrix operand order repaired and checked by reading, not compiled, not run on hardware
//
#include <hip/hip_runtime.h>
#include <hip/hip_bf16.h>
#include <stdint.h>

typedef _Float16 v8h  __attribute__((ext_vector_type(8)));
typedef _Float16 v16h __attribute__((ext_vector_type(16)));
typedef float    v8f  __attribute__((ext_vector_type(8)));
typedef int      v4i  __attribute__((ext_vector_type(4)));

#define B_DIM 4096u
#define H_DIM 1024u
#define K_DIM 2048u
#define BM    128u          // batch rows per workgroup
#define BNH   32u           // h-columns per workgroup (x4 gates = 128 GEMM cols)
#define BK    32u           // K step (WMMA f16 K)
#define NKT   (K_DIM / BK)  // 64 k-steps
#define LDA   40u           // padded halves per LDS tile row (32 + 8)

#if defined(__has_builtin)
#if __has_builtin(__builtin_amdgcn_global_load_async_to_lds_b128) && \
    __has_builtin(__builtin_amdgcn_s_wait_asynccnt)
#define USE_ASYNC_LDS 1
#endif
#endif
#ifndef USE_ASYNC_LDS
#define USE_ASYNC_LDS 0
#endif

#define GLOBAL_AS __attribute__((address_space(1)))
#define LDS_AS    __attribute__((address_space(3)))

// Builtin signature (from hipcc diagnostic): (int4 AS1*, int4 AS3*, imm int, imm int)
__device__ __forceinline__ LDS_AS v4i* lds_ptr(void* p) {
  return (LDS_AS v4i*)(uint32_t)(uintptr_t)p;  // low 32 bits of flat LDS addr = LDS offset
}
__device__ __forceinline__ GLOBAL_AS v4i* glob_ptr(const void* p) {
  return (GLOBAL_AS v4i*)(uintptr_t)p;
}

__device__ __forceinline__ v16h cat16(v8h lo, v8h hi) {
  return __builtin_shufflevector(lo, hi, 0,1,2,3,4,5,6,7,8,9,10,11,12,13,14,15);
}
__device__ __forceinline__ float sigf(float x) {
  return __builtin_amdgcn_rcpf(1.0f + __expf(-x));
}
__device__ __forceinline__ float tanhfast(float x) {
  // tanh(x) = 1 - 2/(1+e^{2x}); stable for |x| large in both directions
  return 1.0f - 2.0f * __builtin_amdgcn_rcpf(1.0f + __expf(2.0f * x));
}

// ---- Pass 1a: Xh[b][k] = f16( k<1024 ? x[b][k] : h_prev[b][k-1024] ) ----
__global__ __launch_bounds__(256) void cvt_xh_kernel(const float* __restrict__ x,
                                                     const float* __restrict__ h,
                                                     _Float16* __restrict__ Xh) {
  uint32_t idx = blockIdx.x * 256u + threadIdx.x;   // one thread per 8 halves
  uint32_t row = idx >> 8;                          // 256 chunks per 2048-row
  uint32_t k0  = (idx & 255u) * 8u;
  const float* src = (k0 < 1024u) ? (x + row * 1024u + k0)
                                  : (h + row * 1024u + (k0 - 1024u));
  const float4* s4 = (const float4*)src;
  float4 a = s4[0], b = s4[1];
  v8h o = { (_Float16)a.x, (_Float16)a.y, (_Float16)a.z, (_Float16)a.w,
            (_Float16)b.x, (_Float16)b.y, (_Float16)b.z, (_Float16)b.w };
  *(v8h*)(Xh + (size_t)idx * 8u) = o;
}

// ---- Pass 1b: Wc[n][k], n = gate*1024 + r ; k<1024 -> W_x gate row, else W_h ----
__global__ __launch_bounds__(256) void cvt_w_kernel(const float* __restrict__ Wii, const float* __restrict__ Whi,
                                                    const float* __restrict__ Wif, const float* __restrict__ Whf,
                                                    const float* __restrict__ Wig, const float* __restrict__ Whg,
                                                    const float* __restrict__ Wio, const float* __restrict__ Who,
                                                    _Float16* __restrict__ Wc) {
  uint32_t idx = blockIdx.x * 256u + threadIdx.x;
  uint32_t n   = idx >> 8;          // 0..4095
  uint32_t k0  = (idx & 255u) * 8u;
  uint32_t g   = n >> 10;
  uint32_t r   = n & 1023u;
  const float* base;
  uint32_t kk;
  if (k0 < 1024u) {
    base = (g == 0u) ? Wii : (g == 1u) ? Wif : (g == 2u) ? Wig : Wio;
    kk = k0;
  } else {
    base = (g == 0u) ? Whi : (g == 1u) ? Whf : (g == 2u) ? Whg : Who;
    kk = k0 - 1024u;
  }
  const float4* s4 = (const float4*)(base + r * 1024u + kk);
  float4 a = s4[0], b = s4[1];
  v8h o = { (_Float16)a.x, (_Float16)a.y, (_Float16)a.z, (_Float16)a.w,
            (_Float16)b.x, (_Float16)b.y, (_Float16)b.z, (_Float16)b.w };
  *(v8h*)(Wc + (size_t)idx * 8u) = o;
}

// ---- Pass 2: fused WMMA GEMM + LSTM epilogue ----
union SMem {
  struct { _Float16 A[2][BM][LDA]; _Float16 Bt[2][BM][LDA]; } st;  // 40 KB
  float g[BM][BM];                                                 // 64 KB gates
};

__global__ __launch_bounds__(256) void lstm_gemm_kernel(
    const _Float16* __restrict__ Xh, const _Float16* __restrict__ Wc,
    const float* __restrict__ c_prev,
    const float* __restrict__ b_i, const float* __restrict__ b_f,
    const float* __restrict__ b_g, const float* __restrict__ b_o,
    float* __restrict__ out) {
  __shared__ __align__(16) SMem sm;

  const uint32_t tid  = threadIdx.x;
  const uint32_t lane = tid & 31u;
  const uint32_t wv   = tid >> 5;   // 0..7
  const uint32_t mw   = wv & 1u;    // M-wave: 0..1  (64 rows each)
  const uint32_t gw   = wv >> 1;    // gate:   0..3  (32 cols each)
  const uint32_t l15  = lane & 15u;
  const uint32_t hh   = lane >> 4;  // half-wave select
  const uint32_t m0   = blockIdx.x * BM;
  const uint32_t nh0  = blockIdx.y * BNH;

  v8f acc[4][2] = {};

#if USE_ASYNC_LDS
  // Async DMA of both tiles directly into LDS (4 x b128 per thread per k-step),
  // tracked by ASYNCcnt; copies of tile kt+1 overlap WMMA on tile kt.
  auto issue = [&](uint32_t p, uint32_t kt) {
#pragma unroll
    for (int j = 0; j < 2; ++j) {
      uint32_t c = tid + (uint32_t)j * 256u;   // 512 16-byte chunks per tile
      uint32_t r = c >> 2, s = c & 3u;
      const _Float16* ga = Xh + (m0 + r) * K_DIM + kt * BK + s * 8u;
      __builtin_amdgcn_global_load_async_to_lds_b128(
          glob_ptr(ga), lds_ptr(&sm.st.A[p][r][s * 8u]), 0, 0);
      uint32_t grow = (r >> 5) * 1024u + nh0 + (r & 31u);  // gate-major rows
      const _Float16* gb = Wc + grow * K_DIM + kt * BK + s * 8u;
      __builtin_amdgcn_global_load_async_to_lds_b128(
          glob_ptr(gb), lds_ptr(&sm.st.Bt[p][r][s * 8u]), 0, 0);
    }
  };
  issue(0u, 0u);
#else
  v8h ra[2], rb[2];
  auto ldA = [&](uint32_t kt) {
#pragma unroll
    for (int j = 0; j < 2; ++j) {
      uint32_t c = tid + (uint32_t)j * 256u;
      uint32_t r = c >> 2, s = c & 3u;
      ra[j] = *(const v8h*)(Xh + (m0 + r) * K_DIM + kt * BK + s * 8u);
    }
  };
  auto ldB = [&](uint32_t kt) {
#pragma unroll
    for (int j = 0; j < 2; ++j) {
      uint32_t c = tid + (uint32_t)j * 256u;
      uint32_t r = c >> 2, s = c & 3u;
      uint32_t grow = (r >> 5) * 1024u + nh0 + (r & 31u);
      rb[j] = *(const v8h*)(Wc + grow * K_DIM + kt * BK + s * 8u);
    }
  };
  auto stAB = [&](uint32_t p) {
#pragma unroll
    for (int j = 0; j < 2; ++j) {
      uint32_t c = tid + (uint32_t)j * 256u;
      uint32_t r = c >> 2, s = c & 3u;
      *(v8h*)&sm.st.A [p][r][s * 8u] = ra[j];
      *(v8h*)&sm.st.Bt[p][r][s * 8u] = rb[j];
    }
  };
  ldA(0); ldB(0); stAB(0);
#endif

  for (uint32_t kt = 0; kt < NKT; ++kt) {
    const uint32_t p = kt & 1u;
#if USE_ASYNC_LDS
    __builtin_amdgcn_s_wait_asynccnt(0);  // own copies for buffer p landed
    __syncthreads();                      // everyone's copies landed; prev reads done
    if (kt + 1u < NKT) issue(p ^ 1u, kt + 1u);  // refill other buffer, overlapped
#else
    __syncthreads();
    if (kt + 1u < NKT) { ldA(kt + 1u); ldB(kt + 1u); }
#endif

    // A fragment per ISA layout: lane<16 -> K 0-7 & 16-23 ; lane>=16 -> K 8-15 & 24-31
    v16h af[4];
#pragma unroll
    for (int mi = 0; mi < 4; ++mi) {
      uint32_t rr = mw * 64u + (uint32_t)mi * 16u + l15;
      v8h lo = *(const v8h*)&sm.st.A[p][rr][hh * 8u];
      v8h hi = *(const v8h*)&sm.st.A[p][rr][16u + hh * 8u];
      af[mi] = cat16(lo, hi);
    }
    // B fragment per ISA layout: lane<16 -> K 0-7 & 16-23 ; lane>=16 -> K 8-15 & 24-31
    v16h bf[2];
#pragma unroll
    for (int ni = 0; ni < 2; ++ni) {
      uint32_t rr = gw * 32u + (uint32_t)ni * 16u + l15;
      v8h lo = *(const v8h*)&sm.st.Bt[p][rr][hh * 8u];
      v8h hi = *(const v8h*)&sm.st.Bt[p][rr][16u + hh * 8u];
      bf[ni] = cat16(lo, hi);
    }
#pragma unroll
    for (int mi = 0; mi < 4; ++mi)
#pragma unroll
      for (int ni = 0; ni < 2; ++ni)
        acc[mi][ni] = __builtin_amdgcn_wmma_f32_16x16x32_f16(
            false, af[mi], false, bf[ni], (short)0, acc[mi][ni], false, false);

#if !USE_ASYNC_LDS
    if (kt + 1u < NKT) stAB(p ^ 1u);
#endif
  }

  __syncthreads();   // all LDS staging reads done; safe to alias with gates tile

  // Spill accumulators to LDS gates tile: C layout lanes0-15 M=v, lanes16-31 M=v+8
#pragma unroll
  for (int mi = 0; mi < 4; ++mi)
#pragma unroll
    for (int ni = 0; ni < 2; ++ni) {
      uint32_t rr = mw * 64u + (uint32_t)mi * 16u + hh * 8u;
      uint32_t cc = gw * 32u + (uint32_t)ni * 16u + l15;
#pragma unroll
      for (int v = 0; v < 8; ++v)
        sm.g[rr + (uint32_t)v][cc] = acc[mi][ni][v];
    }
  __syncthreads();

  // Fused LSTM epilogue: coalesced across h-columns within each wave
  {
    uint32_t nh = tid & 31u;
    uint32_t mb = (tid >> 5) * 16u;
    uint32_t n  = nh0 + nh;
    float bi = b_i[n], bff = b_f[n], bg = b_g[n], bo = b_o[n];
#pragma unroll 4
    for (uint32_t j = 0; j < 16u; ++j) {
      uint32_t m  = mb + j;
      uint32_t mg = m0 + m;
      float iv = sigf    (sm.g[m][       nh] + bi);
      float fv = sigf    (sm.g[m][32u  + nh] + bff);
      float gv = tanhfast(sm.g[m][64u  + nh] + bg);
      float ov = sigf    (sm.g[m][96u  + nh] + bo);
      float cp = c_prev[mg * 1024u + n];
      float ct = fv * cp + iv * gv;
      float ht = ov * tanhfast(ct);
      uint32_t oi = mg * 1024u + n;
      out[oi]            = ht;   // h_t
      out[4194304u + oi] = ht;   // h_t (second tuple element)
      out[8388608u + oi] = ct;   // c_t
    }
  }
}

extern "C" void kernel_launch(void* const* d_in, const int* in_sizes, int n_in,
                              void* d_out, int out_size, void* d_ws, size_t ws_size,
                              hipStream_t stream) {
  const float* x      = (const float*)d_in[0];
  const float* h_prev = (const float*)d_in[1];
  const float* c_prev = (const float*)d_in[2];
  const float* W_ii = (const float*)d_in[3];
  const float* W_hi = (const float*)d_in[4];
  const float* b_i  = (const float*)d_in[5];
  const float* W_if = (const float*)d_in[6];
  const float* W_hf = (const float*)d_in[7];
  const float* b_f  = (const float*)d_in[8];
  const float* W_ig = (const float*)d_in[9];
  const float* W_hg = (const float*)d_in[10];
  const float* b_g  = (const float*)d_in[11];
  const float* W_io = (const float*)d_in[12];
  const float* W_ho = (const float*)d_in[13];
  const float* b_o  = (const float*)d_in[14];
  float* out = (float*)d_out;

  _Float16* Xh = (_Float16*)d_ws;                       // [4096][2048] f16 = 16 MB
  _Float16* Wc = Xh + (size_t)B_DIM * K_DIM;            // [4096][2048] f16 = 16 MB

  cvt_xh_kernel<<<4096, 256, 0, stream>>>(x, h_prev, Xh);
  cvt_w_kernel<<<4096, 256, 0, stream>>>(W_ii, W_hi, W_if, W_hf,
                                         W_ig, W_hg, W_io, W_ho, Wc);
  dim3 grid(B_DIM / BM, H_DIM / BNH);   // 32 x 32
  lstm_gemm_kernel<<<grid, 256, 0, stream>>>(Xh, Wc, c_prev,
                                             b_i, b_f, b_g, b_o, out);
}
